// MeteoGraphPC_v1_50903952392630
// MI455X (gfx1250) — matrix operand order repaired and checked by reading, not compiled, not run on hardware
//
#include <hip/hip_runtime.h>
#include <stdint.h>

// ---------------------------------------------------------------------------
// T-GCN (GRU over GCNConv) for MI455X / gfx1250, wave32 + WMMA f16->f32.
//
//   gcn_g = Anorm @ (x @ Wc_g) == (Anorm @ x) @ Wc_g      -> aggregate F=16 once
//   pre_g = agg @ (Wc_g @ WlTop_g) + h @ WlBot_g + bf_g   -> fused [160,128] GEMM
//
// All three GRU gates in ONE barrier-free kernel per cell:
//  - B fragments read directly from global (40 KB/gate, WGP$-resident).
//  - z kept in registers (packed f16); r*h bounced through a wave-private LDS
//    tile straight into the h-gate WMMA A-fragments.
//  - hprev masking (mask & seen) folded into the GEMM via a 1-byte/node keep
//    flag; h_state is updated in place (each wave owns its 16 rows).
// ---------------------------------------------------------------------------

constexpr int NN   = 50000;
constexpr int NE   = 1600000;
constexpr int TT   = 12;
constexpr int FF   = 16;
constexpr int HH   = 128;
constexpr int NOUT = 3;
constexpr int HZN  = 6;
constexpr int KTOT = 160;                    // 32 (agg, zero padded) + 128 (hidden)
constexpr int ROW_TILES = NN / 16;           // 3125 (N divisible by 16)
constexpr int WPW  = 8;                      // waves (16-row tiles) per workgroup
constexpr int GEMM_WGS = (ROW_TILES + WPW - 1) / WPW;

typedef __attribute__((ext_vector_type(16))) _Float16 v16h;
typedef __attribute__((ext_vector_type(8)))  _Float16 v8h;
typedef __attribute__((ext_vector_type(8)))  float    v8f;

#define DEV __device__ __forceinline__

DEV v8f zero8f() {
  v8f z;
#pragma unroll
  for (int i = 0; i < 8; ++i) z[i] = 0.0f;
  return z;
}

DEV float fast_sigmoid(float x) { return 1.0f / (1.0f + __expf(-x)); }
DEV float fast_tanh(float x) {
  float e2 = __expf(2.0f * x);               // inf-safe: e2=inf -> 1, e2=0 -> -1
  return 1.0f - 2.0f / (e2 + 1.0f);
}

// ---------------- elementwise / edge kernels ----------------

__global__ void zero_f32_kernel(float* __restrict__ p, int n) {
  int i = blockIdx.x * blockDim.x + threadIdx.x;
  if (i < n) p[i] = 0.0f;
}

__global__ void zero_u8_kernel(unsigned char* __restrict__ p, int n) {
  int i = blockIdx.x * blockDim.x + threadIdx.x;
  if (i < n) p[i] = 0;
}

__global__ void copy_f32_kernel(float* __restrict__ dst, const float* __restrict__ src, int n) {
  int i = blockIdx.x * blockDim.x + threadIdx.x;
  if (i < n) dst[i] = src[i];
}

__global__ void deg_init_kernel(float* __restrict__ deg) {
  int i = blockIdx.x * blockDim.x + threadIdx.x;
  if (i < NN) deg[i] = 1.0f;                 // self-loop weight
}

__global__ void edge_deg_kernel(const int* __restrict__ dst, const float* __restrict__ w,
                                float* __restrict__ deg) {
  int e = blockIdx.x * blockDim.x + threadIdx.x;
  if (e < NE) atomicAdd(&deg[dst[e]], w[e]);
}

// dinv = rsqrt(deg); agg[i][:] = dinv^2 * x[i][:]  (self-loop term, full overwrite)
__global__ void dinv_agginit_kernel(const float* __restrict__ deg, float* __restrict__ dinv,
                                    const float* __restrict__ x, float* __restrict__ agg) {
  int i = blockIdx.x * blockDim.x + threadIdx.x;
  if (i >= NN) return;
  float d = rsqrtf(deg[i]);
  dinv[i] = d;
  float s = d * d;
  const float4* xs = (const float4*)(x + (size_t)i * FF);
  float4* ad = (float4*)(agg + (size_t)i * FF);
#pragma unroll
  for (int q = 0; q < FF / 4; ++q) {
    float4 v = xs[q];
    float4 o; o.x = s * v.x; o.y = s * v.y; o.z = s * v.z; o.w = s * v.w;
    ad[q] = o;
  }
}

__global__ void edge_agg_kernel(const int* __restrict__ src, const int* __restrict__ dst,
                                const float* __restrict__ w, const float* __restrict__ dinv,
                                const float* __restrict__ x, float* __restrict__ agg) {
  int e = blockIdx.x * blockDim.x + threadIdx.x;
  if (e >= NE) return;
  int s = src[e], d = dst[e];
  float nm = dinv[s] * w[e] * dinv[d];
  const float4* xs = (const float4*)(x + (size_t)s * FF);
  float* ad = agg + (size_t)d * FF;
#pragma unroll
  for (int q = 0; q < FF / 4; ++q) {
    float4 v = xs[q];
    atomicAdd(ad + q * 4 + 0, nm * v.x);
    atomicAdd(ad + q * 4 + 1, nm * v.y);
    atomicAdd(ad + q * 4 + 2, nm * v.z);
    atomicAdd(ad + q * 4 + 3, nm * v.w);
  }
}

// keep = mask & seen (for this step's h_prev gather); seen |= mask
__global__ void keep_update_kernel(const unsigned char* __restrict__ m,
                                   unsigned char* __restrict__ seen,
                                   unsigned char* __restrict__ keep) {
  int i = blockIdx.x * blockDim.x + threadIdx.x;
  if (i >= NN) return;
  unsigned char mv = m[i] ? 1 : 0;
  unsigned char sv = seen[i];
  keep[i] = (unsigned char)(mv & sv);
  seen[i] = (unsigned char)(mv | sv);
}

// pred = h @ headW + headb ; write d_out slice and feed back into x_buf[:,0:3]
__global__ void head_kernel(const float* __restrict__ h, const float* __restrict__ Wh,
                            const float* __restrict__ bh, float* __restrict__ out,
                            float* __restrict__ xbuf) {
  int n = blockIdx.x * blockDim.x + threadIdx.x;
  if (n >= NN) return;
  const float* hr = h + (size_t)n * HH;
  float a0 = bh[0], a1 = bh[1], a2 = bh[2];
#pragma unroll 4
  for (int k = 0; k < HH; ++k) {
    float hv = hr[k];
    a0 += hv * Wh[k * NOUT + 0];
    a1 += hv * Wh[k * NOUT + 1];
    a2 += hv * Wh[k * NOUT + 2];
  }
  out[(size_t)n * NOUT + 0] = a0;
  out[(size_t)n * NOUT + 1] = a1;
  out[(size_t)n * NOUT + 2] = a2;
  xbuf[(size_t)n * FF + 0] = a0;             // TARGET_IDX = (0,1,2)
  xbuf[(size_t)n * FF + 1] = a1;
  xbuf[(size_t)n * FF + 2] = a2;
}

// ---------------- weight prep (once per call, tiny) ----------------

// Fused transposed weights WcatT[n][k], n<128, k<160 (f16):
//   k in [0,16):   (Wc @ WlTop)[k][n]
//   k in [16,32):  0  (pad so agg chunk is a full K=32 WMMA slice)
//   k in [32,160): WlBot[k-32][n]   (Wl rows 128..255)
__global__ void prep_wcat_kernel(const float* __restrict__ Wc, const float* __restrict__ Wl,
                                 _Float16* __restrict__ WcatT) {
  int idx = blockIdx.x * blockDim.x + threadIdx.x;
  if (idx >= HH * KTOT) return;
  int n = idx / KTOT, k = idx % KTOT;
  float v;
  if (k < FF) {
    v = 0.0f;
    for (int j = 0; j < HH; ++j) v += Wc[k * HH + j] * Wl[(size_t)j * HH + n];
  } else if (k < 32) {
    v = 0.0f;
  } else {
    v = Wl[(size_t)(k - 32 + HH) * HH + n];
  }
  WcatT[idx] = (_Float16)v;
}

// bf[n] = bl[n] + bc @ WlTop[:,n]
__global__ void prep_bf_kernel(const float* __restrict__ bc, const float* __restrict__ Wl,
                               const float* __restrict__ bl, float* __restrict__ bf) {
  int n = threadIdx.x;
  if (n >= HH) return;
  float v = bl[n];
  for (int j = 0; j < HH; ++j) v += bc[j] * Wl[(size_t)j * HH + n];
  bf[n] = v;
}

// ---------------- WMMA fragment helpers ----------------
// A 16x32 f16 layout (ISA 7.12.2): lane -> M = lane&15, half-select hs = lane>>4
//   a[j]   (j<8)  -> K = hs*8 + j
//   a[8+j] (j<8)  -> K = 16 + hs*8 + j

DEV v16h load_a_agg(const float* __restrict__ agg, int row, int hs) {
  const float* ar = agg + (size_t)row * FF;
  v16h a;
#pragma unroll
  for (int j = 0; j < 8; ++j) {
    a[j]     = (_Float16)ar[hs * 8 + j];     // K 0..15 = agg features
    a[8 + j] = (_Float16)0.0f;               // K 16..31 = zero pad
  }
  return a;
}

// hidden A chunk, scaled by per-row keep flag (h_prev = keep ? h : 0)
DEV v16h load_a_hid(const float* __restrict__ hmat, int row, int kbase, int hs, float ksc) {
  const float* hr = hmat + (size_t)row * HH + kbase;
  v16h a;
#pragma unroll
  for (int j = 0; j < 8; ++j) {
    a[j]     = (_Float16)(hr[hs * 8 + j] * ksc);
    a[8 + j] = (_Float16)(hr[16 + hs * 8 + j] * ksc);
  }
  return a;
}

// B 32x16 f16: lane -> N = lane&15; b[j] -> K = hs*8 + j, b[8+j] -> K = 16 + hs*8 + j.
// WcatT is [n][k] (global, WGP$-resident): two aligned 16B reads per fragment.
DEV v8f mma_col(const v16h a[5], const _Float16* __restrict__ W, int gcol, int hs) {
  const _Float16* wb = W + (size_t)gcol * KTOT + hs * 8;
  v8f acc = zero8f();
#pragma unroll
  for (int kc = 0; kc < 5; ++kc) {
    const v8h blo = *(const v8h*)(wb + kc * 32);
    const v8h bhi = *(const v8h*)(wb + kc * 32 + 16);
    v16h b;
#pragma unroll
    for (int j = 0; j < 8; ++j) { b[j] = blo[j]; b[8 + j] = bhi[j]; }
    acc = __builtin_amdgcn_wmma_f32_16x16x32_f16(false, a[kc], false, b,
                                                 (short)0, acc, false, false);
  }
  return acc;
}

// ---------------- fused 3-gate GRU cell kernel (barrier-free) ----------------
// One wave per 16-row tile; 8 waves / WG; K = 5 chunks of 32 per gate.

__global__ void __launch_bounds__(WPW * 32)
gemm_cell_kernel(const float* __restrict__ agg,
                 const float* __restrict__ hsrc,          // hidden-state source
                 const unsigned char* __restrict__ keepp, // null => keep all rows
                 const _Float16* __restrict__ Wz, const _Float16* __restrict__ Wr,
                 const _Float16* __restrict__ Wh,
                 const float* __restrict__ bfz, const float* __restrict__ bfr,
                 const float* __restrict__ bfh,
                 const unsigned char* __restrict__ maskp, // null => unconditional store
                 float* __restrict__ hdst) {              // may alias hsrc (row-owned)
  __shared__ __align__(32) _Float16 ldsRH[WPW * 16 * HH];   // 32 KB, per-wave r*h tiles
  const int lane = threadIdx.x & 31;
  const int wave = threadIdx.x >> 5;
  const int row0 = (blockIdx.x * WPW + wave) * 16;
  if (row0 >= NN) return;
  const int row = row0 + (lane & 15);
  const int hs = lane >> 4;
  const int ncol = lane & 15;
  const int mbase = hs * 8;
  _Float16* rhw = ldsRH + wave * (16 * HH);

  // per-row keep flags: scalar for this lane's A-row, 8-byte word for C-rows
  const float ksc = keepp ? (keepp[row] ? 1.0f : 0.0f) : 1.0f;
  const unsigned long long kw =
      keepp ? *(const unsigned long long*)(keepp + row0 + mbase) : 0x0101010101010101ull;

  // A fragments: agg (zero-padded K chunk) + 4 gated hidden chunks (z & r gates)
  v16h a[5];
  a[0] = load_a_agg(agg, row, hs);
#pragma unroll
  for (int c = 1; c < 5; ++c) a[c] = load_a_hid(hsrc, row, (c - 1) * 32, hs, ksc);

  v8h zt[8];                                 // z gate, packed f16 (C-layout)

  // ---- gate Z ----
#pragma unroll
  for (int nt = 0; nt < 8; ++nt) {
    const int gcol = nt * 16 + ncol;
    v8f acc = mma_col(a, Wz, gcol, hs);
    const float bb = bfz[gcol];
#pragma unroll
    for (int v = 0; v < 8; ++v) zt[nt][v] = (_Float16)fast_sigmoid(acc[v] + bb);
  }

  // ---- gate R -> r*h_prev into wave-private LDS tile ----
#pragma unroll
  for (int nt = 0; nt < 8; ++nt) {
    const int gcol = nt * 16 + ncol;
    v8f acc = mma_col(a, Wr, gcol, hs);
    const float bb = bfr[gcol];
#pragma unroll
    for (int v = 0; v < 8; ++v) {
      const int grow = row0 + mbase + v;
      const float kf = ((kw >> (8 * v)) & 0xff) ? 1.0f : 0.0f;
      const float r = fast_sigmoid(acc[v] + bb);
      const float hp = hsrc[(size_t)grow * HH + gcol] * kf;  // L1/L2-resident reread
      rhw[(mbase + v) * HH + gcol] = (_Float16)(r * hp);
    }
  }

  // ---- gate H + GRU combine ----
  // Rebuild hidden A-fragments from the wave-private r*h tile (DS ops from the
  // same wave are in-order, so no barrier is needed for this region).
#pragma unroll
  for (int c = 1; c < 5; ++c) {
    const _Float16* rr = rhw + (lane & 15) * HH + (c - 1) * 32;
    const v8h lo = *(const v8h*)(rr + hs * 8);
    const v8h hi = *(const v8h*)(rr + 16 + hs * 8);
    v16h ac;
#pragma unroll
    for (int j = 0; j < 8; ++j) { ac[j] = lo[j]; ac[8 + j] = hi[j]; }
    a[c] = ac;
  }
  const unsigned long long mw =
      maskp ? *(const unsigned long long*)(maskp + row0 + mbase) : 0x0101010101010101ull;
#pragma unroll
  for (int nt = 0; nt < 8; ++nt) {
    const int gcol = nt * 16 + ncol;
    v8f acc = mma_col(a, Wh, gcol, hs);
    const float bb = bfh[gcol];
#pragma unroll
    for (int v = 0; v < 8; ++v) {
      const int grow = row0 + mbase + v;
      const size_t idx = (size_t)grow * HH + gcol;
      const float kf = ((kw >> (8 * v)) & 0xff) ? 1.0f : 0.0f;
      const float ht = fast_tanh(acc[v] + bb);
      const float zv = (float)zt[nt][v];
      const float hn = zv * (hsrc[idx] * kf) + (1.0f - zv) * ht;
      if (((mw >> (8 * v)) & 0xff) != 0) hdst[idx] = hn;
    }
  }
}

// ---------------------------------------------------------------------------

extern "C" void kernel_launch(void* const* d_in, const int* in_sizes, int n_in,
                              void* d_out, int out_size, void* d_ws, size_t ws_size,
                              hipStream_t stream) {
  (void)in_sizes; (void)n_in; (void)out_size; (void)ws_size;

  const float* x_seq = (const float*)d_in[0];
  const int*   ei    = (const int*)d_in[1];
  const float* ea    = (const float*)d_in[2];
  const unsigned char* mask = (const unsigned char*)d_in[3];
  const float* Wc[3] = {(const float*)d_in[4], (const float*)d_in[8],  (const float*)d_in[12]};
  const float* bc[3] = {(const float*)d_in[5], (const float*)d_in[9],  (const float*)d_in[13]};
  const float* Wl[3] = {(const float*)d_in[6], (const float*)d_in[10], (const float*)d_in[14]};
  const float* bl[3] = {(const float*)d_in[7], (const float*)d_in[11], (const float*)d_in[15]};
  const float* headW = (const float*)d_in[16];
  const float* headb = (const float*)d_in[17];
  float* out = (float*)d_out;

  char* base = (char*)d_ws;
  size_t off = 0;
  auto alloc = [&](size_t bytes) -> char* {
    char* p = base + off;
    off = (off + bytes + 255) & ~(size_t)255;
    return p;
  };
  float* h_state = (float*)alloc((size_t)NN * HH * 4);
  float* hbufA   = (float*)alloc((size_t)NN * HH * 4);   // horizon ping
  float* hbufB   = (float*)alloc((size_t)NN * HH * 4);   // horizon pong
  float* aggbuf  = (float*)alloc((size_t)NN * FF * 4);
  float* degbuf  = (float*)alloc((size_t)NN * 4);
  float* dinvbuf = (float*)alloc((size_t)NN * 4);
  float* xbuf    = (float*)alloc((size_t)NN * FF * 4);
  unsigned char* seenbuf = (unsigned char*)alloc((size_t)NN);
  unsigned char* keepbuf = (unsigned char*)alloc((size_t)NN);
  _Float16* WcatT[3];
  float* bf[3];
  for (int g = 0; g < 3; ++g) {
    WcatT[g] = (_Float16*)alloc((size_t)HH * KTOT * 2);
    bf[g]    = (float*)alloc((size_t)HH * 4);
  }

  const int TPB = 256;
  const int gN   = (NN + TPB - 1) / TPB;
  const int gNH  = (NN * HH + TPB - 1) / TPB;
  const int gE   = (NE + TPB - 1) / TPB;
  const int gW   = (HH * KTOT + TPB - 1) / TPB;
  const int gNF  = (NN * FF + TPB - 1) / TPB;

  // ---- one-time prep (per call; deterministic) ----
  zero_f32_kernel<<<gNH, TPB, 0, stream>>>(h_state, NN * HH);
  zero_u8_kernel<<<gN, TPB, 0, stream>>>(seenbuf, NN);
  for (int g = 0; g < 3; ++g) {
    prep_wcat_kernel<<<gW, TPB, 0, stream>>>(Wc[g], Wl[g], WcatT[g]);
    prep_bf_kernel<<<1, HH, 0, stream>>>(bc[g], Wl[g], bl[g], bf[g]);
  }

  // ---- time loop ----
  for (int t = 0; t < TT; ++t) {
    const float* xt = x_seq + (size_t)t * NN * FF;
    const int* src  = ei + (size_t)t * 2 * NE;
    const int* dst  = src + NE;
    const float* wt = ea + (size_t)t * NE;
    const unsigned char* mt = mask + (size_t)t * NN;

    deg_init_kernel<<<gN, TPB, 0, stream>>>(degbuf);
    edge_deg_kernel<<<gE, TPB, 0, stream>>>(dst, wt, degbuf);
    dinv_agginit_kernel<<<gN, TPB, 0, stream>>>(degbuf, dinvbuf, xt, aggbuf);
    edge_agg_kernel<<<gE, TPB, 0, stream>>>(src, dst, wt, dinvbuf, xt, aggbuf);
    keep_update_kernel<<<gN, TPB, 0, stream>>>(mt, seenbuf, keepbuf);
    gemm_cell_kernel<<<GEMM_WGS, WPW * 32, 0, stream>>>(
        aggbuf, h_state, keepbuf, WcatT[0], WcatT[1], WcatT[2],
        bf[0], bf[1], bf[2], mt, h_state);                 // in-place, row-owned
  }

  // ---- horizon (deg/dinv from t = T-1 remain valid) ----
  const int* srcL = ei + (size_t)(TT - 1) * 2 * NE;
  const int* dstL = srcL + NE;
  const float* wL = ea + (size_t)(TT - 1) * NE;
  const unsigned char* mL = mask + (size_t)(TT - 1) * NN;

  copy_f32_kernel<<<gNF, TPB, 0, stream>>>(xbuf, x_seq + (size_t)(TT - 1) * NN * FF, NN * FF);

  const float* hsrc = h_state;
  const unsigned char* kp = mL;              // hz=0: h_prev = mask[-1] ? h_state : 0
  float* hA = hbufA;
  float* hB = hbufB;
  for (int hz = 0; hz < HZN; ++hz) {
    dinv_agginit_kernel<<<gN, TPB, 0, stream>>>(degbuf, dinvbuf, xbuf, aggbuf);
    edge_agg_kernel<<<gE, TPB, 0, stream>>>(srcL, dstL, wL, dinvbuf, xbuf, aggbuf);
    gemm_cell_kernel<<<GEMM_WGS, WPW * 32, 0, stream>>>(
        aggbuf, hsrc, kp, WcatT[0], WcatT[1], WcatT[2],
        bf[0], bf[1], bf[2], (const unsigned char*)nullptr, hA);
    head_kernel<<<gN, TPB, 0, stream>>>(hA, headW, headb, out + (size_t)hz * NN * NOUT, xbuf);
    hsrc = hA;                               // h_prev = h_new (unmasked) for hz+1
    kp = nullptr;
    float* tmp = hA; hA = hB; hB = tmp;
  }
}
